// EdgeConv_76965813944552
// MI455X (gfx1250) — matrix operand order repaired and checked by reading, not compiled, not run on hardware
//
#include <hip/hip_runtime.h>
#include <hip/hip_bf16.h>

typedef _Float16 v4h  __attribute__((ext_vector_type(4)));
typedef _Float16 v8h  __attribute__((ext_vector_type(8)));
typedef _Float16 v16h __attribute__((ext_vector_type(16)));
typedef float    v4f  __attribute__((ext_vector_type(4)));
typedef float    v8f  __attribute__((ext_vector_type(8)));

#define IN_C   64
#define EDGE_C 32
#define OUT_C  64
#define D_IN   160            // 2*IN_C + EDGE_C
#define NWAVES 4
#define TPB    (NWAVES * 32)

__device__ __forceinline__ v16h make_v16(v8h lo, v8h hi) {
  return __builtin_shufflevector(lo, hi, 0,1,2,3,4,5,6,7,8,9,10,11,12,13,14,15);
}

// ---------------------------------------------------------------------------
// Edge-MLP + scatter-add.  One wave processes one 16-edge tile per iteration.
// Layer1: [16x160] x [160x64]  (5 K-steps x 4 N-tiles WMMA f16->f32)
// Layer2: [16x64]  x [64x64]   (2 K-steps x 4 N-tiles WMMA f16->f32)
// bias2 is algebraically folded into the final mean-division kernel.
// ---------------------------------------------------------------------------
__global__ __launch_bounds__(TPB) void edgeconv_mlp_scatter(
    const float* __restrict__ x,      // [N, 64]
    const int*   __restrict__ ei,     // [2, E]
    const float* __restrict__ ea,     // [E, 32]
    const float* __restrict__ W1,     // [160, 64]
    const float* __restrict__ b1,     // [64]
    const float* __restrict__ W2,     // [64, 64]
    float* __restrict__ out_sum,      // [N, 64] (accumulated)
    float* __restrict__ cnt,          // [N]
    int N, int E)
{
  __shared__ __align__(32) _Float16 sW1t[OUT_C * D_IN];   // transposed [n][k]
  __shared__ __align__(32) _Float16 sW2t[OUT_C * OUT_C];  // transposed [n][k]
  __shared__ __align__(32) _Float16 sMsg[NWAVES][16 * D_IN];
  __shared__ __align__(32) _Float16 sH  [NWAVES][16 * OUT_C];
  __shared__ int sIdx[NWAVES][32];    // [0..15]=row(clamped), [16..31]=col(raw)

  const int tid  = threadIdx.x;
  const int lane = tid & 31;
  const int wv   = tid >> 5;

  // ---- convert + transpose weights into LDS (f16) ----
  for (int i = tid; i < D_IN * OUT_C; i += TPB) {
    int k = i >> 6, n = i & 63;                 // W1[k][n] row-major
    sW1t[n * D_IN + k] = (_Float16)W1[i];
  }
  for (int i = tid; i < OUT_C * OUT_C; i += TPB) {
    int k = i >> 6, n = i & 63;
    sW2t[n * OUT_C + k] = (_Float16)W2[i];
  }
  __syncthreads();

  const int nlow = lane & 15;         // N (and A-row) index within tile
  const int hi16 = (lane >> 4) & 1;   // lane group
  const int moff = hi16 * 8;          // C/D row offset for this lane group

  // ---- persistent B fragments for layer 2 ----
  v16h w2f[2][4];
  #pragma unroll
  for (int ko = 0; ko < 2; ++ko)
    #pragma unroll
    for (int no = 0; no < 4; ++no) {
      int col = no * 16 + nlow;
      int kb  = ko * 32 + hi16 * 16;
      w2f[ko][no] = *(const v16h*)&sW2t[col * OUT_C + kb];
    }

  // ---- per-lane layer-1 biases ----
  float bias1v[4];
  #pragma unroll
  for (int no = 0; no < 4; ++no) bias1v[no] = b1[no * 16 + nlow];

  _Float16* msg = sMsg[wv];
  _Float16* Hs  = sH[wv];
  int*      idx = sIdx[wv];

  const int nTiles = (E + 15) >> 4;
  const int wid    = blockIdx.x * NWAVES + wv;
  const int gw     = gridDim.x * NWAVES;

  const int chunk = lane & 15;        // float4 chunk within a 64-float row
  const int esub  = lane >> 4;        // which of 2 edges this lane group covers

  for (int t = wid; t < nTiles; t += gw) {
    const int e0 = t << 4;
    const bool full = (e0 + 16 <= E);  // wave-uniform

    // prefetch next tile's (coalesced) edge_attr block into cache
    {
      long pe = (long)(e0) + (long)gw * 16;
      if (pe < E) __builtin_prefetch(&ea[pe * EDGE_C], 0, 1);
    }

    // -- edge indices: lanes 0-15 -> clamped row, lanes 16-31 -> raw col --
    {
      int e  = e0 + nlow;
      int rv = (e < E) ? ei[hi16 * E + e] : -1;
      idx[lane] = hi16 ? rv : (rv < 0 ? 0 : rv);
    }
    __builtin_amdgcn_wave_barrier();  // same-wave LDS is in-order (DScnt)

    // ---- gather x[row]: 16 edges x 16 float4 (batched b128 loads) ----
    v4f tR[8];
    #pragma unroll
    for (int it = 0; it < 8; ++it) {
      int e = it * 2 + esub;
      int r = idx[e];
      tR[it] = *(const v4f*)&x[(size_t)r * IN_C + 4 * chunk];
    }
    #pragma unroll
    for (int it = 0; it < 8; ++it) {
      int e = it * 2 + esub;
      v4h h;
      #pragma unroll
      for (int j = 0; j < 4; ++j) h[j] = (_Float16)tR[it][j];
      *(v4h*)&msg[e * D_IN + 4 * chunk] = h;
    }

    // ---- gather x[col] ----
    v4f tC[8];
    #pragma unroll
    for (int it = 0; it < 8; ++it) {
      int e = it * 2 + esub;
      int c = idx[16 + e];
      c = c < 0 ? 0 : c;
      tC[it] = *(const v4f*)&x[(size_t)c * IN_C + 4 * chunk];
    }
    #pragma unroll
    for (int it = 0; it < 8; ++it) {
      int e = it * 2 + esub;
      v4h h;
      #pragma unroll
      for (int j = 0; j < 4; ++j) h[j] = (_Float16)tC[it][j];
      *(v4h*)&msg[e * D_IN + IN_C + 4 * chunk] = h;
    }

    // ---- edge_attr: fully contiguous 16x32 floats = 128 float4 ----
    v4f tE[4];
    #pragma unroll
    for (int it = 0; it < 4; ++it) {
      int q = it * 32 + lane;            // [0,128)
      long ge = (long)e0 + (q >> 3);
      if (ge > (long)E - 1) ge = (long)E - 1;
      tE[it] = *(const v4f*)&ea[ge * EDGE_C + 4 * (q & 7)];
    }
    #pragma unroll
    for (int it = 0; it < 4; ++it) {
      int q = it * 32 + lane;
      int e = q >> 3, c = q & 7;
      v4h h;
      #pragma unroll
      for (int j = 0; j < 4; ++j) h[j] = (_Float16)tE[it][j];
      *(v4h*)&msg[e * D_IN + 2 * IN_C + 4 * c] = h;
    }
    __builtin_amdgcn_wave_barrier();

    // -- A fragments (16x32 f16 layout: K in [c,c+16) contiguous) --
    v16h af[5];
    #pragma unroll
    for (int ko = 0; ko < 5; ++ko) {
      int base = nlow * D_IN + ko * 32 + hi16 * 16;
      v8h lo = *(const v8h*)&msg[base];
      v8h hi = *(const v8h*)&msg[base + 8];
      af[ko] = make_v16(lo, hi);
    }

    // -- layer 1: H = relu(msg @ W1 + b1) --
    #pragma unroll
    for (int no = 0; no < 4; ++no) {
      v16h bf[5];                      // preload all K-step B frags
      #pragma unroll
      for (int ko = 0; ko < 5; ++ko)
        bf[ko] = *(const v16h*)&sW1t[(no * 16 + nlow) * D_IN + ko * 32 + hi16 * 16];
      v8f acc = {};
      #pragma unroll
      for (int ko = 0; ko < 5; ++ko)
        acc = __builtin_amdgcn_wmma_f32_16x16x32_f16(
                  false, af[ko], false, bf[ko], (short)0, acc, false, false);
      #pragma unroll
      for (int r = 0; r < 8; ++r) {
        float h = acc[r] + bias1v[no];
        h = h > 0.0f ? h : 0.0f;
        Hs[(r + moff) * OUT_C + no * 16 + nlow] = (_Float16)h;
      }
    }
    __builtin_amdgcn_wave_barrier();

    // -- A fragments for layer 2 from H tile --
    v16h a2[2];
    #pragma unroll
    for (int ko = 0; ko < 2; ++ko) {
      int base = nlow * OUT_C + ko * 32 + hi16 * 16;
      v8h lo = *(const v8h*)&Hs[base];
      v8h hi = *(const v8h*)&Hs[base + 8];
      a2[ko] = make_v16(lo, hi);
    }

    // -- scatter targets (hoisted out of the no-loop) --
    int cm[8];
    #pragma unroll
    for (int r = 0; r < 8; ++r) cm[r] = idx[16 + r + moff];

    // -- layer 2 + scatter-add into out_sum[col] (bias2 folded out) --
    #pragma unroll
    for (int no = 0; no < 4; ++no) {
      v8f acc = {};
      acc = __builtin_amdgcn_wmma_f32_16x16x32_f16(
                false, a2[0], false, w2f[0][no], (short)0, acc, false, false);
      acc = __builtin_amdgcn_wmma_f32_16x16x32_f16(
                false, a2[1], false, w2f[1][no], (short)0, acc, false, false);
      if (full) {
        #pragma unroll
        for (int r = 0; r < 8; ++r)
          atomicAdd(&out_sum[(size_t)cm[r] * OUT_C + no * 16 + nlow], acc[r]);
      } else {
        #pragma unroll
        for (int r = 0; r < 8; ++r)
          if (cm[r] >= 0)
            atomicAdd(&out_sum[(size_t)cm[r] * OUT_C + no * 16 + nlow], acc[r]);
      }
    }

    // -- degree count (one lane per edge) --
    if (hi16 == 0) {
      int c = idx[16 + nlow];
      if (full || c >= 0) atomicAdd(&cnt[c < 0 ? 0 : c], 1.0f);
    }
  }
}

// ---------------------------------------------------------------------------
__global__ void edgeconv_zero(float* __restrict__ out, float* __restrict__ cnt,
                              int N) {
  int i = blockIdx.x * blockDim.x + threadIdx.x;
  if (i < N * OUT_C) out[i] = 0.0f;
  if (i < N)         cnt[i] = 0.0f;
}

// out = out_sum / max(count,1) + (count>0 ? b2 : 0)
__global__ void edgeconv_div(float* __restrict__ out, const float* __restrict__ cnt,
                             const float* __restrict__ b2, int N) {
  int i = blockIdx.x * blockDim.x + threadIdx.x;
  if (i < N * OUT_C) {
    float c  = cnt[i >> 6];
    float cc = c > 1.0f ? c : 1.0f;        // clip(count, 1)
    float o  = out[i] / cc;
    if (c > 0.0f) o += b2[i & 63];
    out[i] = o;
  }
}

// ---------------------------------------------------------------------------
extern "C" void kernel_launch(void* const* d_in, const int* in_sizes, int n_in,
                              void* d_out, int out_size, void* d_ws, size_t ws_size,
                              hipStream_t stream) {
  (void)n_in; (void)out_size; (void)ws_size;
  const float* x  = (const float*)d_in[0];
  const int*   ei = (const int*)  d_in[1];
  const float* ea = (const float*)d_in[2];
  const float* W1 = (const float*)d_in[3];
  const float* b1 = (const float*)d_in[4];
  const float* W2 = (const float*)d_in[5];
  const float* b2 = (const float*)d_in[6];
  float* out = (float*)d_out;
  float* cnt = (float*)d_ws;          // N floats of scratch

  const int N = in_sizes[0] / IN_C;
  const int E = in_sizes[1] / 2;
  const int total = N * OUT_C;

  edgeconv_zero<<<(total + 255) / 256, 256, 0, stream>>>(out, cnt, N);

  const int nTiles = (E + 15) >> 4;
  int blocks = (nTiles + NWAVES - 1) / NWAVES;
  if (blocks > 2048) blocks = 2048;
  edgeconv_mlp_scatter<<<blocks, TPB, 0, stream>>>(x, ei, ea, W1, b1, W2,
                                                   out, cnt, N, E);

  edgeconv_div<<<(total + 255) / 256, 256, 0, stream>>>(out, cnt, b2, N);
}
